// msaLMMixin_77936476553340
// MI455X (gfx1250) — matrix operand order repaired and checked by reading, not compiled, not run on hardware
//
#include <hip/hip_runtime.h>

// ---------------------------------------------------------------------------
// CDNA5 (gfx1250) fused multimodal attention:
//   k1: QKV projection GEMM  (fp32 x / W -> bf16 Q(scaled), K, V^T)  [WMMA bf16]
//   k2: flash attention; block = 8 waves on 8 consecutive q-tiles of one (b,h);
//       K/V chunks staged in LDS via global_load_async_to_lds_b128, double-
//       buffered against ASYNCcnt; analytic segment mask with chunk skipping
//   k3: output projection GEMM (bf16 attn-out x fp32 Wo -> fp32 out) [WMMA bf16]
// ---------------------------------------------------------------------------

typedef __attribute__((ext_vector_type(16))) __bf16 v16bf;
typedef __attribute__((ext_vector_type(8)))  __bf16 v8bf;
typedef __attribute__((ext_vector_type(8)))  float  v8f;

#define WMMA_BF16(a, b, c) \
  __builtin_amdgcn_wmma_f32_16x16x32_bf16(false, (a), false, (b), (short)0, (c), false, false)

__device__ __forceinline__ unsigned short f2bf(float f) {
  unsigned int u = __float_as_uint(f);
  u += 0x7FFFu + ((u >> 16) & 1u);          // round-to-nearest-even
  return (unsigned short)(u >> 16);
}
__device__ __forceinline__ unsigned int pack_bf2(float lo, float hi) {
  return (unsigned int)f2bf(lo) | ((unsigned int)f2bf(hi) << 16);
}
__device__ __forceinline__ v8f zero8() {
  v8f z = {0.f, 0.f, 0.f, 0.f, 0.f, 0.f, 0.f, 0.f};
  return z;
}

// Build a v16bf fragment from two 16-byte chunks.
__device__ __forceinline__ v16bf load_frag2(const unsigned short* p0,
                                            const unsigned short* p1) {
  v8bf lo = *(const v8bf*)p0;
  v8bf hi = *(const v8bf*)p1;
  return __builtin_shufflevector(lo, hi, 0, 1, 2, 3, 4, 5, 6, 7,
                                 8, 9, 10, 11, 12, 13, 14, 15);
}
// A fragment (16x32 bf16): lane holds row l%16; chunks at K = 8*hs and 16+8*hs.
__device__ __forceinline__ v16bf load_a_frag(const unsigned short* row, int hs) {
  return load_frag2(row + hs * 8, row + 16 + hs * 8);
}
// B fragment (32x16 bf16) from row-major B^T: lane holds col l%16;
// chunks at K = 8*hs and 16+8*hs.
__device__ __forceinline__ v16bf load_b_frag(const unsigned short* rowT, int hs) {
  return load_frag2(rowT + hs * 8, rowT + 16 + hs * 8);
}

// Generic LDS pointer -> 32-bit LDS offset: flat LDS addresses keep the
// offset in addr[31:0] (ISA 10.2 aperture mapping), so truncation suffices.
__device__ __forceinline__ unsigned lds_lo32(const void* p) {
  return (unsigned)(unsigned long long)p;
}
// Async DMA: 16B per lane, global -> LDS, tracked by ASYNCcnt.
__device__ __forceinline__ void async_ld_b128(unsigned lds, const void* g) {
  asm volatile("global_load_async_to_lds_b128 %0, %1, off"
               :: "v"(lds), "v"(g) : "memory");
}

// ---------------------------------------------------------------------------
// GEMM: C[4096x1024] = A[4096x1024] * W[1024x1024]
// modes 0/1/2: A = x (fp32), write Q(scaled bf16)/K(bf16)/V^T(bf16) per-head
// mode  3    : A = attn-out (bf16), write fp32 d_out
// Block tile 128x64, BK=32, 8 waves (4Mx2N), each wave 32x32 via 4 WMMAs.
// ---------------------------------------------------------------------------
#define GD   1024
#define BM   128
#define BN   64
#define BK   32
#define ASTR 40   // halfs; 80B rows -> 16B-aligned b128 fragment loads
#define BSTR 40

__global__ __launch_bounds__(256)
void wmma_gemm_kernel(const float* __restrict__ Af32,
                      const unsigned short* __restrict__ Abf,
                      const float* __restrict__ Wq,
                      const float* __restrict__ Wk,
                      const float* __restrict__ Wv,
                      const float* __restrict__ Wo,
                      unsigned short* __restrict__ Qo,
                      unsigned short* __restrict__ Ko,
                      unsigned short* __restrict__ Vto,
                      float* __restrict__ Fo,
                      int modeBase) {
  __shared__ unsigned short As[BM * ASTR];   // 10240 B
  __shared__ unsigned short Bs[BN * BSTR];   //  5120 B  (stores W^T tile)

  const int mode = modeBase + (int)blockIdx.z;
  const float* W = (mode == 0) ? Wq : (mode == 1) ? Wk : (mode == 2) ? Wv : Wo;

  const int tid  = threadIdx.x;
  const int lane = tid & 31;
  const int wave = tid >> 5;
  const int fm   = lane & 15;
  const int hs   = lane >> 4;
  const int wm   = (wave & 3) * 32;
  const int wn   = (wave >> 2) * 32;
  const int mBase = blockIdx.x * BM;
  const int nBase = blockIdx.y * BN;

  v8f acc00 = zero8(), acc01 = zero8(), acc10 = zero8(), acc11 = zero8();

  for (int k0 = 0; k0 < GD; k0 += BK) {
    // Speculative prefetch of the next BK-step tiles (global_prefetch_b8).
    if (k0 + BK < GD) {
      {
        int row = tid >> 3, col = (tid & 7) << 2;
        if (mode < 3)
          __builtin_prefetch(Af32 + (size_t)(mBase + row) * GD + k0 + BK + col, 0, 3);
        else
          __builtin_prefetch(Abf + (size_t)(mBase + row) * GD + k0 + BK + col, 0, 3);
      }
      {
        int kr = tid >> 4, col = (tid & 15) << 2;
        __builtin_prefetch(W + (size_t)(k0 + BK + kr) * GD + nBase + col, 0, 3);
      }
    }
    // Stage A tile (128x32) as bf16 in LDS.
#pragma unroll
    for (int i = 0; i < 4; ++i) {
      int c   = tid + i * 256;
      int row = c >> 3;
      int col = (c & 7) << 2;
      unsigned int u0, u1;
      if (mode < 3) {
        const float4 v = *(const float4*)(Af32 + (size_t)(mBase + row) * GD + k0 + col);
        u0 = pack_bf2(v.x, v.y);
        u1 = pack_bf2(v.z, v.w);
      } else {
        const uint2 v = *(const uint2*)(Abf + (size_t)(mBase + row) * GD + k0 + col);
        u0 = v.x; u1 = v.y;
      }
      *(uint2*)(As + row * ASTR + col) = make_uint2(u0, u1);
    }
    // Stage W tile transposed (Bs[n][k]) as bf16 in LDS.
#pragma unroll
    for (int i = 0; i < 2; ++i) {
      int c   = tid + i * 256;
      int kr  = c >> 4;
      int col = (c & 15) << 2;
      const float4 v = *(const float4*)(W + (size_t)(k0 + kr) * GD + nBase + col);
      Bs[(col + 0) * BSTR + kr] = f2bf(v.x);
      Bs[(col + 1) * BSTR + kr] = f2bf(v.y);
      Bs[(col + 2) * BSTR + kr] = f2bf(v.z);
      Bs[(col + 3) * BSTR + kr] = f2bf(v.w);
    }
    __syncthreads();

    const unsigned short* a0p = As + (wm + fm) * ASTR;
    const unsigned short* a1p = As + (wm + 16 + fm) * ASTR;
    v16bf a0 = load_a_frag(a0p, hs);
    v16bf a1 = load_a_frag(a1p, hs);
    const unsigned short* b0p = Bs + (wn + fm) * BSTR;
    const unsigned short* b1p = Bs + (wn + 16 + fm) * BSTR;
    v16bf b0 = load_b_frag(b0p, hs);
    v16bf b1 = load_b_frag(b1p, hs);

    acc00 = WMMA_BF16(a0, b0, acc00);
    acc01 = WMMA_BF16(a0, b1, acc01);
    acc10 = WMMA_BF16(a1, b0, acc10);
    acc11 = WMMA_BF16(a1, b1, acc11);
    __syncthreads();
  }

  // Epilogue. C/D layout: col = lane%16, row = vgprIdx + 8*(lane/16).
  const int H = 16, Lh = 2048, Dh = 64;
#pragma unroll
  for (int ti = 0; ti < 2; ++ti) {
#pragma unroll
    for (int tj = 0; tj < 2; ++tj) {
      v8f acc = (ti == 0) ? (tj == 0 ? acc00 : acc01)
                          : (tj == 0 ? acc10 : acc11);
#pragma unroll
      for (int r = 0; r < 8; ++r) {
        int m = mBase + wm + ti * 16 + r + 8 * hs;
        int n = nBase + wn + tj * 16 + fm;
        float val = acc[r];
        if (mode == 3) {
          Fo[(size_t)m * GD + n] = val;
        } else {
          int b = m >> 11, l = m & 2047;
          int h = n >> 6,  dh = n & 63;
          if (mode == 0)
            Qo[(((size_t)(b * H + h)) * Lh + l) * Dh + dh] = f2bf(val * 0.125f);
          else if (mode == 1)
            Ko[(((size_t)(b * H + h)) * Lh + l) * Dh + dh] = f2bf(val);
          else
            Vto[(((size_t)(b * H + h)) * Dh + dh) * Lh + l] = f2bf(val);
        }
      }
    }
  }
}

// ---------------------------------------------------------------------------
// Flash attention.
// Block = (b, h, 128-query stripe); 8 waves = 8 consecutive 16-query tiles.
// Mask: seg = idx>>9. allowed(q,k) = (seg q==seg k && k<=q) || (seg k==0 && seg q!=0).
// 32-key chunks never straddle 512-boundaries; chunk set is block-uniform.
// K/V chunks are staged in LDS by the async DMA path, double-buffered.
// ---------------------------------------------------------------------------
#define KSTR 72   // halfs; 144B rows, 16B-aligned fragment loads
#define VSTR 40
#define PSTR 40

// Next live 32-key chunk for the block (block-uniform).
__device__ __forceinline__ int next_chunk(int kb, int sq, int qmaxB) {
  kb += 32;
  if (sq == 0) return (kb <= qmaxB) ? kb : -1;
  if (kb < 512) return kb;                // rest of the full text segment
  if (kb < sq * 512) kb = sq * 512;       // jump over NO_ATTEND segments
  return (kb <= qmaxB) ? kb : -1;         // causal part of own segment
}

__global__ __launch_bounds__(256)
void flash_attn_kernel(const unsigned short* __restrict__ Q,
                       const unsigned short* __restrict__ K,
                       const unsigned short* __restrict__ Vt,
                       unsigned short* __restrict__ Oa) {
  __shared__ unsigned short Ks[2][32 * KSTR];   // 2 x 4608 B
  __shared__ unsigned short Vs[2][64 * VSTR];   // 2 x 5120 B
  __shared__ unsigned short Pl[8][16 * PSTR];   // 10240 B (per-wave P tiles)

  const int L = 2048, Dh = 64, H = 16;
  const int tid  = threadIdx.x;
  const int lane = tid & 31;
  const int wave = tid >> 5;
  const int fm   = lane & 15;
  const int hs   = lane >> 4;

  const int tile0 = blockIdx.x & 15;          // 16 stripes of 128 queries
  const int h     = (blockIdx.x >> 4) & 15;
  const int b     = blockIdx.x >> 8;

  const size_t bh = (size_t)(b * H + h);
  const unsigned short* Qb = Q  + bh * (size_t)L * Dh;   // [L, Dh] (pre-scaled)
  const unsigned short* Kb = K  + bh * (size_t)L * Dh;   // [L, Dh]
  const unsigned short* Vb = Vt + bh * (size_t)Dh * L;   // [Dh, L]

  const int q0    = tile0 * 128 + wave * 16;  // this wave's query tile
  const int sq    = tile0 >> 2;               // segment (uniform across block)
  const int qmaxB = tile0 * 128 + 127;        // block causal horizon

  // Per-thread async staging assignment: 16B per lane.
  const int krow = tid >> 3, kcol = (tid & 7) << 3;   // K chunk: 32 x 64 halfs
  const int vrow = tid >> 2, vcol = (tid & 3) << 3;   // V chunk: 64 x 32 halfs

  const unsigned short* qrow = Qb + (size_t)(q0 + fm) * Dh;
  v16bf qa0 = load_a_frag(qrow, hs);        // d 0..31
  v16bf qa1 = load_a_frag(qrow + 32, hs);   // d 32..63

  v8f o0 = zero8(), o1 = zero8(), o2 = zero8(), o3 = zero8();
  float mrow[8], lrow[8];
#pragma unroll
  for (int r = 0; r < 8; ++r) { mrow[r] = -3.0e38f; lrow[r] = 0.0f; }

  auto stage = [&](int kb, int pb) {
    async_ld_b128(lds_lo32(&Ks[pb][krow * KSTR + kcol]),
                  Kb + (size_t)(kb + krow) * Dh + kcol);
    async_ld_b128(lds_lo32(&Vs[pb][vrow * VSTR + vcol]),
                  Vb + (size_t)vrow * L + kb + vcol);
  };

  int kb = 0;   // first live chunk is always 0 (text attends causally; others full)
  int pb = 0;
  stage(kb, 0);

  while (kb >= 0) {
    const int kn = next_chunk(kb, sq, qmaxB);
    if (kn >= 0) {
      stage(kn, pb ^ 1);                               // overlap next fill
      asm volatile("s_wait_asynccnt 0x2" ::: "memory"); // current buffer done
    } else {
      asm volatile("s_wait_asynccnt 0x0" ::: "memory");
    }
    __syncthreads();

    const int  sk   = kb >> 9;
    const bool diag = (sk == sq);   // only diagonal-segment chunks need elem mask

    // S = Q K^T for 16 queries x 32 keys (two 16-key subtiles) from LDS.
    const unsigned short* kr0 = &Ks[pb][fm * KSTR];
    const unsigned short* kr1 = &Ks[pb][(16 + fm) * KSTR];
    v16bf kf00 = load_b_frag(kr0, hs);        // keys [kb,kb+16), d 0..31
    v16bf kf01 = load_b_frag(kr0 + 32, hs);   //                  d 32..63
    v16bf kf10 = load_b_frag(kr1, hs);
    v16bf kf11 = load_b_frag(kr1 + 32, hs);

    v8f s0 = zero8(), s1 = zero8();
    s0 = WMMA_BF16(qa0, kf00, s0);
    s0 = WMMA_BF16(qa1, kf01, s0);
    s1 = WMMA_BF16(qa0, kf10, s1);
    s1 = WMMA_BF16(qa1, kf11, s1);

    float sv0[8], sv1[8], cm[8];
#pragma unroll
    for (int r = 0; r < 8; ++r) {
      const int qg  = q0 + r + 8 * hs;
      const int kg0 = kb + fm;
      const int kg1 = kb + 16 + fm;
      sv0[r] = s0[r] + ((!diag || kg0 <= qg) ? 0.0f : -1.0e9f);
      sv1[r] = s1[r] + ((!diag || kg1 <= qg) ? 0.0f : -1.0e9f);
      cm[r]  = fmaxf(sv0[r], sv1[r]);
    }
    // Row reductions stay inside each 16-lane half (masks 1/2/4/8).
#pragma unroll
    for (int off = 1; off < 16; off <<= 1)
#pragma unroll
      for (int r = 0; r < 8; ++r)
        cm[r] = fmaxf(cm[r], __shfl_xor(cm[r], off, 32));

    float scale[8], p0v[8], p1v[8], cs[8];
#pragma unroll
    for (int r = 0; r < 8; ++r) {
      float nm = fmaxf(mrow[r], cm[r]);
      scale[r] = __expf(mrow[r] - nm);
      mrow[r]  = nm;
      p0v[r]   = __expf(sv0[r] - nm);
      p1v[r]   = __expf(sv1[r] - nm);
      cs[r]    = p0v[r] + p1v[r];
    }
#pragma unroll
    for (int off = 1; off < 16; off <<= 1)
#pragma unroll
      for (int r = 0; r < 8; ++r)
        cs[r] += __shfl_xor(cs[r], off, 32);

#pragma unroll
    for (int r = 0; r < 8; ++r) {
      lrow[r] = lrow[r] * scale[r] + cs[r];
      o0[r] *= scale[r]; o1[r] *= scale[r];
      o2[r] *= scale[r]; o3[r] *= scale[r];
    }

    // D-layout -> A-layout transpose of P through this wave's LDS region.
    unsigned short* pw = &Pl[wave][0];
#pragma unroll
    for (int r = 0; r < 8; ++r) {
      const int m = r + 8 * hs;
      pw[m * PSTR + fm]      = f2bf(p0v[r]);
      pw[m * PSTR + 16 + fm] = f2bf(p1v[r]);
    }
    __builtin_amdgcn_wave_barrier();   // same-wave DS ops are in-order
    const unsigned short* prow = pw + fm * PSTR;
    v16bf pa = load_a_frag(prow, hs);

    // O += P V from the staged V tile (rows = dh, cols = chunk-local k).
    v16bf vf0 = load_b_frag(&Vs[pb][(0  + fm) * VSTR], hs);
    v16bf vf1 = load_b_frag(&Vs[pb][(16 + fm) * VSTR], hs);
    v16bf vf2 = load_b_frag(&Vs[pb][(32 + fm) * VSTR], hs);
    v16bf vf3 = load_b_frag(&Vs[pb][(48 + fm) * VSTR], hs);

    o0 = WMMA_BF16(pa, vf0, o0);
    o1 = WMMA_BF16(pa, vf1, o1);
    o2 = WMMA_BF16(pa, vf2, o2);
    o3 = WMMA_BF16(pa, vf3, o3);

    __syncthreads();   // LDS fence: buffer free before it is re-staged
    kb = kn;
    pb ^= 1;
  }

  // Normalize and write attention output as bf16 [B, L, D] for the Wo GEMM.
#pragma unroll
  for (int j = 0; j < 4; ++j) {
    v8f oj = (j == 0) ? o0 : (j == 1) ? o1 : (j == 2) ? o2 : o3;
#pragma unroll
    for (int r = 0; r < 8; ++r) {
      const int qg = q0 + r + 8 * hs;
      const int dh = j * 16 + fm;
      Oa[((size_t)b * L + qg) * 1024 + h * Dh + dh] = f2bf(oj[r] / lrow[r]);
    }
  }
}

// ---------------------------------------------------------------------------
extern "C" void kernel_launch(void* const* d_in, const int* in_sizes, int n_in,
                              void* d_out, int out_size, void* d_ws, size_t ws_size,
                              hipStream_t stream) {
  (void)in_sizes; (void)n_in; (void)out_size; (void)ws_size;
  const float* x  = (const float*)d_in[0];
  const float* Wq = (const float*)d_in[1];
  const float* Wk = (const float*)d_in[2];
  const float* Wv = (const float*)d_in[3];
  const float* Wo = (const float*)d_in[4];
  float* out = (float*)d_out;

  // Workspace: Q | K | V^T | attn-out, each 2*16*2048*64 bf16 = 8 MB (32 MB total).
  unsigned short* ws = (unsigned short*)d_ws;
  const size_t perT = (size_t)2 * 16 * 2048 * 64;
  unsigned short* Qw  = ws;
  unsigned short* Kw  = ws + perT;
  unsigned short* Vtw = ws + 2 * perT;
  unsigned short* Oaw = ws + 3 * perT;

  dim3 blk(256);
  dim3 g1(4096 / BM, 1024 / BN, 3);   // z: Q / K / V
  wmma_gemm_kernel<<<g1, blk, 0, stream>>>(x, nullptr, Wq, Wk, Wv, Wo,
                                           Qw, Kw, Vtw, nullptr, 0);
  dim3 g2(512);                       // (b, h, 128-query stripe)
  flash_attn_kernel<<<g2, blk, 0, stream>>>(Qw, Kw, Vtw, Oaw);
  dim3 g3(4096 / BM, 1024 / BN, 1);   // out = attn @ Wo (fp32 out)
  wmma_gemm_kernel<<<g3, blk, 0, stream>>>(nullptr, Oaw, Wq, Wk, Wv, Wo,
                                           nullptr, nullptr, nullptr, out, 3);
}
